// MultiheadVanillaAttention_71236327571792
// MI455X (gfx1250) — matrix operand order repaired and checked by reading, not compiled, not run on hardware
//
#include <hip/hip_runtime.h>

typedef __attribute__((ext_vector_type(16))) _Float16 v16h;
typedef __attribute__((ext_vector_type(8)))  float    v8f;

#define S_LEN 2048
#define BATCH 2
#define EMB   1024
#define HEADS 16
#define HDIM  64
#define QSTR  (3 * EMB)            // fused QKV row stride
#define NEGBIG (-1.0e30f)

// ---------------------------------------------------------------------------
// CDNA5 async global->LDS copy (ASYNCcnt-tracked, no VGPR data movement).
__device__ __forceinline__ void async_load_b128(const _Float16* gptr, _Float16* lptr) {
  unsigned lds = (unsigned)(size_t)lptr;           // flat LDS aperture: addr[31:0]
  unsigned long long ga = (unsigned long long)(size_t)gptr;
  asm volatile("global_load_async_to_lds_b128 %0, %1, off"
               :: "v"(lds), "v"(ga) : "memory");
}
template <int N>
__device__ __forceinline__ void wait_async() {
  asm volatile("s_wait_asynccnt %0" :: "i"(N) : "memory");
}

// ---------------------------------------------------------------------------
// Assemble a 16-half WMMA fragment from 8 packed-pair dword reads.
// A-operand: b0 = half*8,  b1 = 16 + half*8   (ISA 16-bit A 16x32 layout)
// B-operand: b0 = half*8,  b1 = 16 + half*8   (same k order as A)
__device__ __forceinline__ v16h frag8(const _Float16* p, int b0, int b1) {
  union { unsigned u[8]; v16h h; } f;
#pragma unroll
  for (int v = 0; v < 4; ++v) f.u[v]     = *(const unsigned*)(p + b0 + 2 * v);
#pragma unroll
  for (int v = 0; v < 4; ++v) f.u[4 + v] = *(const unsigned*)(p + b1 + 2 * v);
  return f.h;
}

__global__ void cvt_f32_f16(const float* __restrict__ src, _Float16* __restrict__ dst,
                            int n, float scale) {
  int i = blockIdx.x * blockDim.x + threadIdx.x;
  if (i < n) dst[i] = (_Float16)(src[i] * scale);
}

__global__ void scale_f32(const float* __restrict__ src, float* __restrict__ dst,
                          int n, float scale) {
  int i = blockIdx.x * blockDim.x + threadIdx.x;
  if (i < n) dst[i] = src[i] * scale;
}

// ---------------------------------------------------------------------------
// C = A(MxK,f16) * W(NxK,f16)^T + bias (scaled). Block tile 128x256, 8 waves,
// wave tile 64x64 (16 WMMA / k-step). 3-stage async-to-LDS pipeline, K-chunk 32.
template <int OUTF16>
__global__ __launch_bounds__(256)
void gemm_f16(const _Float16* __restrict__ A,
              const _Float16* __restrict__ W,
              const float* __restrict__ bias,
              void* __restrict__ outp,
              int M, int N, int K, float scale)
{
  __shared__ _Float16 Ash[3][128][40];   // [stage][m][k] row stride 80B (16B aligned)
  __shared__ _Float16 Bsh[3][256][40];   // [stage][n][k] = W rows verbatim

  const int tid  = threadIdx.x;
  const int lane = tid & 31;
  const int wave = tid >> 5;
  const int half = lane >> 4;
  const int ln   = lane & 15;
  const int waveM = wave >> 2;          // 0..1 : 64 rows
  const int waveN = wave & 3;           // 0..3 : 64 cols
  const int mBase = blockIdx.y * 128;
  const int nBase = blockIdx.x * 256;

  v8f acc[4][4];
#pragma unroll
  for (int i = 0; i < 4; ++i)
#pragma unroll
    for (int j = 0; j < 4; ++j) acc[i][j] = {};

  auto stage = [&](int buf, int k0) {
#pragma unroll
    for (int it = 0; it < 2; ++it) {
      int idx = tid + it * 256, row = idx >> 2, part = idx & 3;
      async_load_b128(&A[(size_t)(mBase + row) * K + k0 + part * 8], &Ash[buf][row][part * 8]);
    }
#pragma unroll
    for (int it = 0; it < 4; ++it) {
      int idx = tid + it * 256, row = idx >> 2, part = idx & 3;
      async_load_b128(&W[(size_t)(nBase + row) * K + k0 + part * 8], &Bsh[buf][row][part * 8]);
    }
  };

  const int nIter = K / 32;
  stage(0, 0);
  for (int i = 0; i < nIter; ++i) {
    const int cur = i % 3;
    if (i + 1 < nIter) {
      stage((i + 1) % 3, (i + 1) * 32);
      wait_async<6>();                 // oldest 6 (stage i) have landed
    } else {
      wait_async<0>();
    }
    __syncthreads();

    v16h a[4];
#pragma unroll
    for (int mt = 0; mt < 4; ++mt)
      a[mt] = frag8(&Ash[cur][waveM * 64 + mt * 16 + ln][0], half * 8, 16 + half * 8);
#pragma unroll
    for (int nt = 0; nt < 4; ++nt) {
      v16h bf = frag8(&Bsh[cur][waveN * 64 + nt * 16 + ln][0], half * 8, 16 + half * 8);
#pragma unroll
      for (int mt = 0; mt < 4; ++mt)
        acc[mt][nt] = __builtin_amdgcn_wmma_f32_16x16x32_f16(
            false, a[mt], false, bf, (short)0, acc[mt][nt], false, false);
    }
  }

#pragma unroll
  for (int nt = 0; nt < 4; ++nt) {
    int col = nBase + waveN * 64 + nt * 16 + ln;
    float bv = bias[col];
#pragma unroll
    for (int mt = 0; mt < 4; ++mt)
#pragma unroll
      for (int r = 0; r < 8; ++r) {
        int row = mBase + waveM * 64 + mt * 16 + r + half * 8;
        float v = (acc[mt][nt][r] + bv) * scale;
        if (OUTF16) ((_Float16*)outp)[(size_t)row * N + col] = (_Float16)v;
        else        ((float*)outp)[(size_t)row * N + col] = v;
      }
  }
}

// ---------------------------------------------------------------------------
// Flash-attention (causal). Block = 128 query rows of one (b,h); 8 waves x 16 rows.
// Q/K/V live in the fused (S*B) x 3E buffer (row stride QSTR); ctx uses stride E.
__global__ __launch_bounds__(256)
void attn_kernel(const _Float16* __restrict__ Q,   // pre-scaled by D^-0.5 (via W,b)
                 const _Float16* __restrict__ Kk,
                 const _Float16* __restrict__ V,
                 _Float16* __restrict__ ctx)
{
  __shared__ _Float16 Ksh[128][72];      // [key][d]   (B-operand layout for QK^T)
  __shared__ _Float16 Vsh[64][136];      // [d][key]   (B-operand layout for PV)
  __shared__ _Float16 Psh[8][16][136];   // per-wave P (A-operand layout)

  const int tid  = threadIdx.x;
  const int lane = tid & 31;
  const int w    = tid >> 5;
  const int half = lane >> 4;
  const int ln   = lane & 15;
  const int qBase = blockIdx.x * 128;
  const int b = blockIdx.y / HEADS;
  const int h = blockIdx.y % HEADS;
  const size_t headOff = (size_t)h * HDIM;

  const int rowMin = qBase + w * 16;
  const int rowMax = rowMin + 15;
  const _Float16* qp = Q + ((size_t)(rowMin + ln) * BATCH + b) * QSTR + headOff;
  v16h Qa[2];
#pragma unroll
  for (int c = 0; c < 2; ++c)
    Qa[c] = frag8(qp, c * 32 + half * 8, c * 32 + 16 + half * 8);

  v8f acc[4];
#pragma unroll
  for (int j = 0; j < 4; ++j) acc[j] = {};
  float m[8], l[8];
#pragma unroll
  for (int r = 0; r < 8; ++r) { m[r] = NEGBIG; l[r] = 0.0f; }

  for (int kb = 0; kb < qBase + 128; kb += 128) {
    // K tile via async copies (overlaps with the V transpose staging below)
#pragma unroll
    for (int it = 0; it < 4; ++it) {
      int idx = tid + it * 256, row = idx >> 3, part = idx & 7;
      async_load_b128(&Kk[((size_t)(kb + row) * BATCH + b) * QSTR + headOff + part * 8],
                      &Ksh[row][part * 8]);
    }
    // V tile transposed -> [d][key]
#pragma unroll
    for (int it = 0; it < 16; ++it) {
      int idx = tid + it * 256, key = idx >> 5, dp = idx & 31;
      union { unsigned u; _Float16 hh[2]; } t;
      t.u = *(const unsigned*)&V[((size_t)(kb + key) * BATCH + b) * QSTR + headOff + dp * 2];
      Vsh[dp * 2 + 0][key] = t.hh[0];
      Vsh[dp * 2 + 1][key] = t.hh[1];
    }
    wait_async<0>();
    __syncthreads();

    v8f s[8];
#pragma unroll
    for (int nt = 0; nt < 8; ++nt) {
      int keyMin = kb + nt * 16;
      if (keyMin > rowMax) {              // wave-uniform: EXEC stays all-1s for WMMA
#pragma unroll
        for (int r = 0; r < 8; ++r) s[nt][r] = NEGBIG;
        continue;
      }
      v8f sc = {};
#pragma unroll
      for (int c = 0; c < 2; ++c) {
        v16h kf = frag8(&Ksh[nt * 16 + ln][0], c * 32 + half * 8, c * 32 + 16 + half * 8);
        sc = __builtin_amdgcn_wmma_f32_16x16x32_f16(false, Qa[c], false, kf, (short)0, sc, false, false);
      }
      if (keyMin + 15 > rowMin) {
        int kj = keyMin + ln;
#pragma unroll
        for (int r = 0; r < 8; ++r)
          if (kj > rowMin + r + half * 8) sc[r] = NEGBIG;
      }
      s[nt] = sc;
    }

#pragma unroll
    for (int r = 0; r < 8; ++r) {
      float t = s[0][r];
#pragma unroll
      for (int nt = 1; nt < 8; ++nt) t = fmaxf(t, s[nt][r]);
#pragma unroll
      for (int off = 1; off < 16; off <<= 1) t = fmaxf(t, __shfl_xor(t, off, 32));
      float newm = fmaxf(m[r], t);
      float corr = __expf(m[r] - newm);
      m[r] = newm;
      float sum = 0.0f;
#pragma unroll
      for (int nt = 0; nt < 8; ++nt) {
        float pv = __expf(s[nt][r] - newm);
        s[nt][r] = pv;
        sum += pv;
      }
#pragma unroll
      for (int off = 1; off < 16; off <<= 1) sum += __shfl_xor(sum, off, 32);
      l[r] = l[r] * corr + sum;
#pragma unroll
      for (int j = 0; j < 4; ++j) acc[j][r] *= corr;
    }

#pragma unroll
    for (int nt = 0; nt < 8; ++nt)
#pragma unroll
      for (int r = 0; r < 8; ++r)
        Psh[w][r + half * 8][nt * 16 + ln] = (_Float16)s[nt][r];

#pragma unroll
    for (int kc = 0; kc < 4; ++kc) {
      if (kb + kc * 32 > rowMax) break;
      v16h pa = frag8(&Psh[w][ln][0], kc * 32 + half * 8, kc * 32 + 16 + half * 8);
#pragma unroll
      for (int j = 0; j < 4; ++j) {
        v16h vf = frag8(&Vsh[j * 16 + ln][0], kc * 32 + half * 8, kc * 32 + 16 + half * 8);
        acc[j] = __builtin_amdgcn_wmma_f32_16x16x32_f16(false, pa, false, vf, (short)0, acc[j], false, false);
      }
    }
    __syncthreads();
  }

#pragma unroll
  for (int r = 0; r < 8; ++r) {
    float inv = 1.0f / l[r];
    int qi = rowMin + r + half * 8;
    _Float16* op = ctx + ((size_t)qi * BATCH + b) * EMB + headOff;
#pragma unroll
    for (int j = 0; j < 4; ++j)
      op[j * 16 + ln] = (_Float16)(acc[j][r] * inv);
  }
}

// ---------------------------------------------------------------------------
extern "C" void kernel_launch(void* const* d_in, const int* in_sizes, int n_in,
                              void* d_out, int out_size, void* d_ws, size_t ws_size,
                              hipStream_t stream)
{
  const float* x  = (const float*)d_in[0];
  const float* wq = (const float*)d_in[1];
  const float* bq = (const float*)d_in[2];
  const float* wk = (const float*)d_in[3];
  const float* bk = (const float*)d_in[4];
  const float* wv = (const float*)d_in[5];
  const float* bv = (const float*)d_in[6];
  const float* wo = (const float*)d_in[7];
  const float* bo = (const float*)d_in[8];

  const int M = S_LEN * BATCH;  // 4096 tokens
  const int K = EMB;            // 1024

  _Float16* p     = (_Float16*)d_ws;
  _Float16* xh    = p; p += (size_t)M * K;
  _Float16* qkvh  = p; p += (size_t)M * QSTR;
  _Float16* ch    = p; p += (size_t)M * EMB;
  _Float16* wqkvh = p; p += (size_t)QSTR * K;   // rows: [wq | wk | wv]
  _Float16* woh   = p; p += (size_t)EMB * K;
  float*    bqkv  = (float*)p;                  // 3072 f32

  const float scaling = 0.125f;  // D^-0.5, baked into wq/bq
  int nx = M * K;
  int nw = EMB * K;
  cvt_f32_f16<<<nx / 256, 256, 0, stream>>>(x,  xh, nx, 1.0f);
  cvt_f32_f16<<<nw / 256, 256, 0, stream>>>(wq, wqkvh,              nw, scaling);
  cvt_f32_f16<<<nw / 256, 256, 0, stream>>>(wk, wqkvh + (size_t)nw, nw, 1.0f);
  cvt_f32_f16<<<nw / 256, 256, 0, stream>>>(wv, wqkvh + 2*(size_t)nw, nw, 1.0f);
  cvt_f32_f16<<<nw / 256, 256, 0, stream>>>(wo, woh, nw, 1.0f);
  scale_f32<<<4, 256, 0, stream>>>(bq, bqkv,        EMB, scaling);
  scale_f32<<<4, 256, 0, stream>>>(bk, bqkv + EMB,  EMB, 1.0f);
  scale_f32<<<4, 256, 0, stream>>>(bv, bqkv + 2*EMB, EMB, 1.0f);

  // fused QKV projection: (4096 x 1024) x (3072 x 1024)^T -> 4096 x 3072
  dim3 gqkv(QSTR / 256, M / 128);               // 12 x 32 = 384 blocks
  gemm_f16<1><<<gqkv, 256, 0, stream>>>(xh, wqkvh, bqkv, qkvh, M, QSTR, K, 1.0f);

  dim3 ga(S_LEN / 128, BATCH * HEADS);          // 16 x 32 = 512 blocks
  attn_kernel<<<ga, 256, 0, stream>>>(qkvh, qkvh + EMB, qkvh + 2 * EMB, ch);

  dim3 go(EMB / 256, M / 128);                  // 4 x 32 = 128 blocks
  gemm_f16<0><<<go, 256, 0, stream>>>(ch, woh, bo, d_out, M, EMB, K, 1.0f);
}
